// RNNClassifier_20925080666402
// MI455X (gfx1250) — matrix operand order repaired and checked by reading, not compiled, not run on hardware
//
#include <hip/hip_runtime.h>
#include <math.h>

// ---------------------------------------------------------------------------
// GRU classifier for MI455X (gfx1250), wave32 + WMMA bf16 16x16x32.
// One persistent workgroup per 16-row batch tile; loops all S=256 steps
// locally (recurrence is batch-independent -> no inter-WG sync).
// Weights pre-converted to bf16 in fragment-swizzled layout in d_ws.
// x[s+1] tile is fetched with GLOBAL_LOAD_ASYNC_TO_LDS_B128 (ASYNCcnt),
// overlapping the HBM/L2 fetch with the WMMA phase of step s.
// ---------------------------------------------------------------------------

typedef __bf16 bf16_t;
typedef __attribute__((ext_vector_type(16))) __bf16 v16bf;
typedef __attribute__((ext_vector_type(8)))  __bf16 v8bf;
typedef __attribute__((ext_vector_type(8)))  float  v8f;

#define B_  512
#define S_  256
#define E_  256
#define H_  512
#define L_  8
#define BT  16          // batch tile per workgroup (one WMMA M tile)

#define KPX (E_ + 8)    // padded LDS row stride (bf16 elems) for x tile
#define KPH (H_ + 8)    // padded LDS row stride (bf16 elems) for h tile

#define SX_BYTES  (16 * KPX * 2)               // 8448
#define SH_BYTES  (16 * KPH * 2)               // 16640
#define XF_OFF    (SX_BYTES + 2 * SH_BYTES)    // 41728 (16B aligned)
#define XF_BYTES  (16 * E_ * 4)                // 16384 fp32 staging tile
#define ARENA_BYTES (XF_OFF + XF_BYTES)        // 58112

static __device__ __forceinline__ v8f wmma_bf16(v16bf a, v16bf b, v8f c) {
  // 8 args: (neg_a, A, neg_b, B, c_mod, C, reuse_a, reuse_b)
  return __builtin_amdgcn_wmma_f32_16x16x32_bf16(
      false, a, false, b, (short)0, c, false, false);
}

// A fragment (16x32, M x K) from LDS row-major tile.
// lane&15 = M row; lane>>4 selects K sub-ranges per ISA layout:
//   VGPR0..3: K = half*8 .. half*8+7 ; VGPR4..7: K = 16+half*8 .. +7
static __device__ __forceinline__ v16bf load_a(const bf16_t* row, int kt, int half) {
  const bf16_t* p = row + kt * 32 + half * 8;
  v8bf lo = *(const v8bf*)(p);
  v8bf hi = *(const v8bf*)(p + 16);
  return __builtin_shufflevector(lo, hi, 0,1,2,3,4,5,6,7,8,9,10,11,12,13,14,15);
}

// B fragment (32x16, K x N) pre-swizzled in global ws: 32 lanes x 16 bf16.
static __device__ __forceinline__ v16bf load_b(const bf16_t* frag, int lane) {
  return *(const v16bf*)(frag + lane * 16);
}

static __device__ __forceinline__ float sigf(float x) {
  return 1.0f / (1.0f + __expf(-x));
}

// ---- CDNA5 async global->LDS copy (ASYNCcnt path), via inline asm ---------
// VDST VGPR holds the byte offset into the wave's LDS allocation; `arena` is
// the only __shared__ object so it sits at LDS offset 0.
static __device__ __forceinline__ void async_ld_b128(unsigned ldsOff, const void* g) {
  asm volatile("global_load_async_to_lds_b128 %0, %1, off"
               :: "v"(ldsOff), "v"((unsigned long long)(uintptr_t)g)
               : "memory");
}

static __device__ __forceinline__ void wait_asynccnt0() {
#if __has_builtin(__builtin_amdgcn_s_wait_asynccnt)
  __builtin_amdgcn_s_wait_asynccnt(0);
#else
  asm volatile("s_wait_asynccnt 0x0" ::: "memory");
#endif
}

// Issue the 16KB fp32 x tile for timestep s into xF (64B per thread).
static __device__ __forceinline__ void issue_x_tile(const float* x, int b0, int s, int tid) {
  const int m  = tid >> 4;               // row 0..15
  const int cb = (tid & 15) * 64;        // byte offset within 1KB row
  const char* src = (const char*)(x + ((size_t)(b0 + m) * S_ + s) * E_) + cb;
  const unsigned dst = XF_OFF + (unsigned)(m * (E_ * 4) + cb);
#pragma unroll
  for (int c = 0; c < 4; ++c)
    async_ld_b128(dst + 16 * c, src + 16 * c);
}

// ---------------------------------------------------------------------------
// Weight fp32 -> bf16 fragment-swizzle convert.
// dst fragment f=(ntile*KT+kt): elem = W[ntile*16 + (lane&15)]
//                                      [kt*32 + (lane>>4)*8 + (e<8 ? e : e+8)]  e=0..15
// stored contiguously: dst[(f*32+lane)*16 + e]  (fully coalesced stores)
// ---------------------------------------------------------------------------
__global__ void cvt_frag(const float* __restrict__ src,
                         unsigned short* __restrict__ dstu,
                         int KT, int K) {
  int gid  = blockIdx.x * blockDim.x + threadIdx.x;
  int frag = gid >> 5;
  int lane = gid & 31;
  int ntile = frag / KT;
  int kt    = frag - ntile * KT;
  int row   = ntile * 16 + (lane & 15);
  int k0    = kt * 32 + (lane >> 4) * 8;
  const float* sp = src + (size_t)row * K + k0;
  bf16_t* dp = (bf16_t*)dstu + (size_t)gid * 16;
#pragma unroll
  for (int e = 0; e < 16; ++e) dp[e] = (bf16_t)sp[(e < 8) ? e : e + 8];
}

// ---------------------------------------------------------------------------
// Persistent GRU kernel: grid = B/BT = 32 blocks, 256 threads (8 waves).
// Wave w owns hidden-unit tiles jt = 4w..4w+3; for each jt it accumulates
// gate columns {jt, 32+jt, 64+jt} (r, z, n) so the nonlinearity runs
// entirely in registers on the D fragments.
// ---------------------------------------------------------------------------
__global__ __launch_bounds__(256)
void gru_persist(const float* __restrict__ x,
                 const unsigned short* __restrict__ wihFu,
                 const unsigned short* __restrict__ whhFu,
                 const float* __restrict__ b_ih,
                 const float* __restrict__ b_hh,
                 const float* __restrict__ W_out,
                 const float* __restrict__ b_out,
                 float* __restrict__ out) {
  __shared__ __align__(16) char arena[ARENA_BYTES];
  bf16_t* sX  = (bf16_t*)(arena);
  bf16_t* sHa = (bf16_t*)(arena + SX_BYTES);
  bf16_t* sHb = (bf16_t*)(arena + SX_BYTES + SH_BYTES);
  const float* xF = (const float*)(arena + XF_OFF);   // async-landed fp32 tile
  float*  hF  = (float*)(arena);                  // reused post-loop: [16][512]
  float*  lg  = (float*)(arena + 16 * 512 * 4);   // reused post-loop: [16][8]

  const bf16_t* wih = (const bf16_t*)wihFu;
  const bf16_t* whh = (const bf16_t*)whhFu;

  const int tid  = threadIdx.x;
  const int w    = tid >> 5;
  const int lane = tid & 31;
  const int half = lane >> 4;
  const int nco  = lane & 15;
  const int b0   = blockIdx.x * BT;

  // kick off the async fetch of x tile for s=0 immediately
  issue_x_tile(x, b0, 0, tid);

  // h0 = 0
  for (int i = tid; i < 16 * KPH; i += 256) sHa[i] = (bf16_t)0.0f;

  // per-lane gate biases (column j = jt*16 + nco), hoisted out of the s-loop
  float bR[4], bZ[4], bIN[4], bHN[4];
#pragma unroll
  for (int q = 0; q < 4; ++q) {
    int j = (w * 4 + q) * 16 + nco;
    bR[q]  = b_ih[j]        + b_hh[j];
    bZ[q]  = b_ih[512 + j]  + b_hh[512 + j];
    bIN[q] = b_ih[1024 + j];
    bHN[q] = b_hh[1024 + j];
  }

  float hreg[4][8];
#pragma unroll
  for (int q = 0; q < 4; ++q)
#pragma unroll
    for (int e = 0; e < 8; ++e) hreg[q][e] = 0.0f;

  bf16_t* cur = sHa;
  bf16_t* nxt = sHb;

  for (int s = 0; s < S_; ++s) {
    wait_asynccnt0();   // x tile s has landed in xF (this wave's issues)
    __syncthreads();    // all waves landed; prior step's LDS reads done

    // convert staged fp32 tile -> bf16 sX (pure LDS->LDS, no global latency)
    {
      int m  = tid >> 4;
      int e0 = (tid & 15) * 16;
      const float* sp = xF + m * E_ + e0;
      bf16_t* dp = sX + m * KPX + e0;
#pragma unroll
      for (int e = 0; e < 16; ++e) dp[e] = (bf16_t)sp[e];
    }
    __syncthreads();    // sX visible; all xF readers done

    // overlap: fetch next step's x tile while we do this step's WMMA work
    if (s + 1 < S_) issue_x_tile(x, b0, s + 1, tid);

    const bf16_t* aRowX = sX  + (lane & 15) * KPX;
    const bf16_t* aRowH = cur + (lane & 15) * KPH;

#pragma unroll
    for (int q = 0; q < 4; ++q) {
      const int jt = w * 4 + q;
      v8f aR, aZ, aIN, aHN;
#pragma unroll
      for (int e = 0; e < 8; ++e) {
        aR[e] = bR[q]; aZ[e] = bZ[q]; aIN[e] = bIN[q]; aHN[e] = bHN[q];
      }

      // input-side: gi = x @ W_ih^T   (K = E = 256, 8 k-tiles)
#pragma unroll 4
      for (int kt = 0; kt < 8; ++kt) {
        v16bf a = load_a(aRowX, kt, half);
        aR  = wmma_bf16(a, load_b(wih + ((size_t)((0 * 32 + jt) * 8 + kt)) * 512, lane), aR);
        aZ  = wmma_bf16(a, load_b(wih + ((size_t)((1 * 32 + jt) * 8 + kt)) * 512, lane), aZ);
        aIN = wmma_bf16(a, load_b(wih + ((size_t)((2 * 32 + jt) * 8 + kt)) * 512, lane), aIN);
      }
      // hidden-side: gh = h @ W_hh^T  (K = H = 512, 16 k-tiles)
#pragma unroll 4
      for (int kt = 0; kt < 16; ++kt) {
        v16bf a = load_a(aRowH, kt, half);
        aR  = wmma_bf16(a, load_b(whh + ((size_t)((0 * 32 + jt) * 16 + kt)) * 512, lane), aR);
        aZ  = wmma_bf16(a, load_b(whh + ((size_t)((1 * 32 + jt) * 16 + kt)) * 512, lane), aZ);
        aHN = wmma_bf16(a, load_b(whh + ((size_t)((2 * 32 + jt) * 16 + kt)) * 512, lane), aHN);
      }

      // gate math fully in-register on the D fragment:
      // lane -> column j, VGPR e -> row m = e + 8*half
      const int j = jt * 16 + nco;
#pragma unroll
      for (int e = 0; e < 8; ++e) {
        float r  = sigf(aR[e]);
        float z  = sigf(aZ[e]);
        float nn = tanhf(aIN[e] + r * aHN[e]);
        float h  = (1.0f - z) * nn + z * hreg[q][e];
        hreg[q][e] = h;
        nxt[(e + 8 * half) * KPH + j] = (bf16_t)h;  // bf16 copy for next step
      }
    }

    bf16_t* t = cur; cur = nxt; nxt = t;
  }

  // ---- classifier head: logits = h @ W_out^T + b_out, softmax over L=8 ----
  __syncthreads();
#pragma unroll
  for (int q = 0; q < 4; ++q) {
    int j = (w * 4 + q) * 16 + nco;
#pragma unroll
    for (int e = 0; e < 8; ++e) hF[(e + 8 * half) * 512 + j] = hreg[q][e];
  }
  __syncthreads();

  if (tid < 128) {
    int m = tid >> 3, l = tid & 7;
    const float* hr = hF + m * 512;
    const float* wr = W_out + l * 512;
    float acc = b_out[l];
#pragma unroll 8
    for (int k = 0; k < 512; ++k) acc += hr[k] * wr[k];
    lg[m * 8 + l] = acc;
  }
  __syncthreads();

  if (tid < 16) {
    int m = tid;
    float mx = lg[m * 8];
#pragma unroll
    for (int l = 1; l < 8; ++l) mx = fmaxf(mx, lg[m * 8 + l]);
    float ex[8], sum = 0.0f;
#pragma unroll
    for (int l = 0; l < 8; ++l) { ex[l] = __expf(lg[m * 8 + l] - mx); sum += ex[l]; }
    float inv = 1.0f / sum;
#pragma unroll
    for (int l = 0; l < 8; ++l) out[(size_t)(b0 + m) * 8 + l] = ex[l] * inv;
  }
}

// ---------------------------------------------------------------------------
extern "C" void kernel_launch(void* const* d_in, const int* in_sizes, int n_in,
                              void* d_out, int out_size, void* d_ws, size_t ws_size,
                              hipStream_t stream) {
  const float* x     = (const float*)d_in[0];   // [B,S,E]
  const float* W_ih  = (const float*)d_in[1];   // [3H,E]
  const float* W_hh  = (const float*)d_in[2];   // [3H,H]
  const float* b_ih  = (const float*)d_in[3];
  const float* b_hh  = (const float*)d_in[4];
  const float* W_out = (const float*)d_in[5];   // [L,H]
  const float* b_out = (const float*)d_in[6];

  // ws layout: bf16 fragment-swizzled weights
  unsigned short* wihF = (unsigned short*)d_ws;            // 96*8 *512 elems
  unsigned short* whhF = wihF + (size_t)96 * 8 * 512;      // 96*16*512 elems

  // W_ih: 96 ntiles x 8 ktiles -> 768 frags * 32 lanes = 24576 threads
  cvt_frag<<<96, 256, 0, stream>>>(W_ih, wihF, 8, E_);
  // W_hh: 96 ntiles x 16 ktiles -> 1536 frags * 32 lanes = 49152 threads
  cvt_frag<<<192, 256, 0, stream>>>(W_hh, whhF, 16, H_);

  gru_persist<<<B_ / BT, 256, 0, stream>>>(x, wihF, whhF, b_ih, b_hh,
                                           W_out, b_out, (float*)d_out);
}
